// MultiheadAttention_52518860096368
// MI455X (gfx1250) — matrix operand order repaired and checked by reading, not compiled, not run on hardware
//
#include <hip/hip_runtime.h>

typedef __bf16 bf16;
typedef __attribute__((ext_vector_type(16))) __bf16 v16bf;
typedef __attribute__((ext_vector_type(8)))  __bf16 v8bf;
typedef __attribute__((ext_vector_type(4)))  __bf16 v4bf;
typedef __attribute__((ext_vector_type(8)))  float  v8f;
typedef __attribute__((ext_vector_type(4)))  float  v4f;

#define WMMA_BF16(A,B,C) \
  __builtin_amdgcn_wmma_f32_16x16x32_bf16(false,(A),false,(B),(short)0,(C),false,false)

constexpr int  B_  = 2;
constexpr int  S_  = 2048;
constexpr int  D_  = 1024;
constexpr int  H_  = 16;
constexpr int  DH  = 64;
constexpr float SCALE = 0.03125f;   // D^-0.5 = 1024^-0.5

// ---------------------------------------------------------------------------
// gfx1250 async global->LDS copy, 16B per lane (ASYNCcnt tracked).
// VDST VGPR holds the wave-relative LDS byte address (= low 32 bits of the
// generic pointer for __shared__ objects per the LDS aperture mapping).
// ---------------------------------------------------------------------------
__device__ __forceinline__ void async_copy16(bf16* lds_dst, const bf16* g)
{
    uint32_t lds = (uint32_t)(uintptr_t)lds_dst;
    uint64_t ga  = (uint64_t)(uintptr_t)g;
    asm volatile("global_load_async_to_lds_b128 %0, %1, off"
                 :: "v"(lds), "v"(ga) : "memory");
}

// ---------------------------------------------------------------------------
// Kernel 0a: plain fp32 -> bf16 cast (query, qk_w, out_w), 4 elems/thread.
// ---------------------------------------------------------------------------
__global__ __launch_bounds__(256) void cast_bf16_kernel(
    const float* __restrict__ src, bf16* __restrict__ dst)
{
    int idx = (blockIdx.x * 256 + threadIdx.x) * 4;
    v4f x = *(const v4f*)(src + idx);
    v4bf y;
#pragma unroll
    for (int i = 0; i < 4; ++i) y[i] = (bf16)x[i];
    *(v4bf*)(dst + idx) = y;
}

// ---------------------------------------------------------------------------
// Kernel 0b: value [B,S,D] fp32 -> V^T [b,h,dh,S] bf16 (per-head transpose so
// P@V B-fragments are contiguous 32B runs).
// ---------------------------------------------------------------------------
__global__ __launch_bounds__(256) void packv_kernel(
    const float* __restrict__ value, bf16* __restrict__ vt)
{
    size_t idx = (size_t)blockIdx.x * blockDim.x + threadIdx.x;   // over B*S*D
    int d  = (int)(idx % D_);
    size_t t = idx / D_;
    int s  = (int)(t % S_);
    int b  = (int)(t / S_);
    int h  = d >> 6;
    int dd = d & 63;
    vt[(((size_t)b * H_ + h) * DH + dd) * S_ + s] = (bf16)value[idx];
}

// ---------------------------------------------------------------------------
// Kernel 1: qk = query @ qk_w^T + qk_b  ->  Q[b,h,S,dh], K[b,h,S,dh] (bf16).
// All-bf16 operands (pre-packed).  One wave = 16(M) x 64(N) tile, K=1024.
// ---------------------------------------------------------------------------
__global__ __launch_bounds__(256) void qk_proj_kernel(
    const bf16* __restrict__ qry, const bf16* __restrict__ w,
    const float* __restrict__ qk_b,
    bf16* __restrict__ qout, bf16* __restrict__ kout)
{
    const int lane = threadIdx.x & 31;
    const int wid  = blockIdx.x * (blockDim.x >> 5) + (threadIdx.x >> 5);
    const int tiles_n = (2 * D_) / 64;        // 32
    const int tm = wid / tiles_n;
    const int tn = wid % tiles_n;
    const int half = lane >> 4;
    const int l16  = lane & 15;
    const int bk = half * 8;
    const int kb = half * 16;

    const bf16* aptr = qry + (size_t)(tm * 16 + l16) * D_;
    const bf16* wptr[4];
#pragma unroll
    for (int t = 0; t < 4; ++t)
        wptr[t] = w + (size_t)(tn * 64 + t * 16 + l16) * D_;

    v8f acc[4];
#pragma unroll
    for (int t = 0; t < 4; ++t)
#pragma unroll
        for (int i = 0; i < 8; ++i) acc[t][i] = 0.f;

    for (int ks = 0; ks < D_ / 32; ++ks) {
        __builtin_prefetch((const void*)(aptr + (ks + 8) * 32), 0, 0);
        __builtin_prefetch((const void*)(wptr[0] + (ks + 8) * 32), 0, 0);
        v8bf lo = *(const v8bf*)(aptr + ks * 32 + kb);
        v8bf hi = *(const v8bf*)(aptr + ks * 32 + kb + 8);
        v16bf a;
#pragma unroll
        for (int i = 0; i < 8; ++i) { a[i] = lo[i]; a[i + 8] = hi[i]; }
#pragma unroll
        for (int t = 0; t < 4; ++t) {
            v16bf bfr = *(const v16bf*)(wptr[t] + ks * 32 + kb);
            acc[t] = WMMA_BF16(a, bfr, acc[t]);
        }
    }

#pragma unroll
    for (int t = 0; t < 4; ++t) {
        int   j    = tn * 64 + t * 16 + l16;
        float bias = qk_b[j];
        bool  isq  = j < D_;
        int   jj   = isq ? j : j - D_;
        int   hh   = jj >> 6;
        int   dd   = jj & 63;
        bf16* base = isq ? qout : kout;
#pragma unroll
        for (int r = 0; r < 8; ++r) {
            int row = tm * 16 + r + 8 * half;
            int bb  = row >> 11;
            int ss  = row & 2047;
            base[(((size_t)bb * H_ + hh) * S_ + ss) * DH + dd] = (bf16)(acc[t][r] + bias);
        }
    }
}

// ---------------------------------------------------------------------------
// Kernel 2: flash attention.  Block = 8 waves = 128 q rows of one (b,h).
// K/V tiles (32 kv) cooperatively staged into LDS with
// global_load_async_to_lds_b128, double buffered, published via barriers.
// ---------------------------------------------------------------------------
__global__ __launch_bounds__(256) void attn_kernel(
    const bf16* __restrict__ q, const bf16* __restrict__ k,
    const bf16* __restrict__ vt, bf16* __restrict__ o)
{
    __shared__ __align__(16) bf16 kbuf[2][32 * 64];   // [kv][dh]
    __shared__ __align__(16) bf16 vbuf[2][64 * 32];   // [dh][kv]
    __shared__ __align__(16) bf16 plds[8][16 * 32];   // per-wave P staging

    const int tid   = threadIdx.x;
    const int lane  = tid & 31;
    const int wslot = tid >> 5;
    const int bh    = blockIdx.x >> 4;      // (b*H + h)
    const int qblk  = blockIdx.x & 15;      // 128-row q block
    const int qt    = qblk * 8 + wslot;     // this wave's 16-row q tile
    const int half  = lane >> 4;
    const int l16   = lane & 15;
    const int bk    = half * 8;
    const int kb    = half * 16;

    const size_t basehd = (size_t)bh * S_ * DH;
    const bf16*  kg0    = k + basehd;                   // K rows for this head
    const bf16*  vg0    = vt + (size_t)bh * DH * S_;    // V^T rows for this head

    // ---- Q A-fragments (register resident for the whole kv sweep) ----------
    const bf16* qrow = q + basehd + (size_t)(qt * 16 + l16) * DH;
    v16bf afrag[2];
#pragma unroll
    for (int ks = 0; ks < 2; ++ks) {
        v8bf lo = *(const v8bf*)(qrow + ks * 32 + kb);
        v8bf hi = *(const v8bf*)(qrow + ks * 32 + kb + 8);
#pragma unroll
        for (int i = 0; i < 8; ++i) { afrag[ks][i] = lo[i]; afrag[ks][i + 8] = hi[i]; }
    }

    v8f acc[4];
    float mrow[8], lrow[8];
#pragma unroll
    for (int t = 0; t < 4; ++t)
#pragma unroll
        for (int i = 0; i < 8; ++i) acc[t][i] = 0.f;
#pragma unroll
    for (int r = 0; r < 8; ++r) { mrow[r] = -1.0e30f; lrow[r] = 0.f; }

    // Cooperative stage of one 32-kv K+V tile: 512 x 16B chunks over 256
    // threads -> each thread issues 2 async copies (2 per wave instruction).
    const int vd = tid >> 2;        // V^T row (dh index) 0..63
    const int vc = tid & 3;         // 16B chunk within row
    auto stage = [&](int buf, int kv) {
        async_copy16(&kbuf[buf][tid * 8], kg0 + (size_t)kv * DH + tid * 8);
        async_copy16(&vbuf[buf][vd * 32 + vc * 8], vg0 + (size_t)vd * S_ + kv + vc * 8);
    };

    stage(0, 0);

    for (int it = 0; it < S_ / 32; ++it) {
        const int buf = it & 1;
        if (it + 1 < S_ / 32) {
            stage(buf ^ 1, (it + 1) * 32);
            asm volatile("s_wait_asynccnt 0x2" ::: "memory");  // older tile done
        } else {
            asm volatile("s_wait_asynccnt 0x0" ::: "memory");
        }
        __syncthreads();            // publish current buffer

        // ---- scores: 16 q rows x 32 kv cols (from LDS K tile) --------------
        v8f s0, s1;
#pragma unroll
        for (int i = 0; i < 8; ++i) { s0[i] = 0.f; s1[i] = 0.f; }
#pragma unroll
        for (int ks = 0; ks < 2; ++ks) {
            v16bf b0 = *(const v16bf*)(&kbuf[buf][l16 * 64 + ks * 32 + kb]);
            v16bf b1 = *(const v16bf*)(&kbuf[buf][(16 + l16) * 64 + ks * 32 + kb]);
            s0 = WMMA_BF16(afrag[ks], b0, s0);
            s1 = WMMA_BF16(afrag[ks], b1, s1);
        }

        // ---- online softmax (row = r + 8*half, col = lane%16 / +16) --------
#pragma unroll
        for (int r = 0; r < 8; ++r) {
            float sm0 = s0[r] * SCALE;
            float sm1 = s1[r] * SCALE;
            float mx  = fmaxf(sm0, sm1);
            mx = fmaxf(mx, __shfl_xor(mx, 1, 32));
            mx = fmaxf(mx, __shfl_xor(mx, 2, 32));
            mx = fmaxf(mx, __shfl_xor(mx, 4, 32));
            mx = fmaxf(mx, __shfl_xor(mx, 8, 32));
            float mn    = fmaxf(mrow[r], mx);
            float alpha = __expf(mrow[r] - mn);
            mrow[r] = mn;
            float p0 = __expf(sm0 - mn);
            float p1 = __expf(sm1 - mn);
            float rs = p0 + p1;
            rs += __shfl_xor(rs, 1, 32);
            rs += __shfl_xor(rs, 2, 32);
            rs += __shfl_xor(rs, 4, 32);
            rs += __shfl_xor(rs, 8, 32);
            lrow[r] = lrow[r] * alpha + rs;
#pragma unroll
            for (int t = 0; t < 4; ++t) acc[t][r] *= alpha;
            int mr = r + 8 * half;
            plds[wslot][mr * 32 + l16]      = (bf16)p0;
            plds[wslot][mr * 32 + 16 + l16] = (bf16)p1;
        }
        asm volatile("s_wait_dscnt 0x0" ::: "memory");

        // ---- P relayout (wave-local LDS -> A fragment, K=32 kv) ------------
        const bf16* pr = &plds[wslot][l16 * 32];
        v8bf plo = *(const v8bf*)(pr + kb);
        v8bf phi = *(const v8bf*)(pr + kb + 8);
        v16bf pf;
#pragma unroll
        for (int i = 0; i < 8; ++i) { pf[i] = plo[i]; pf[i + 8] = phi[i]; }

        // ---- O += P @ V (4 dh column tiles, V from LDS) --------------------
#pragma unroll
        for (int t = 0; t < 4; ++t) {
            v16bf vf = *(const v16bf*)(&vbuf[buf][(t * 16 + l16) * 32 + kb]);
            acc[t] = WMMA_BF16(pf, vf, acc[t]);
        }
        __syncthreads();            // everyone done with this buffer
    }

    // ---- normalize, write merged-head O [B,S,D] bf16 ------------------------
    const int b = bh >> 4, h = bh & 15;
#pragma unroll
    for (int r = 0; r < 8; ++r) {
        float inv  = 1.0f / lrow[r];
        int   srow = qt * 16 + r + 8 * half;
#pragma unroll
        for (int t = 0; t < 4; ++t) {
            int col = h * 64 + t * 16 + l16;
            o[((size_t)b * S_ + srow) * D_ + col] = (bf16)(acc[t][r] * inv);
        }
    }
}

// ---------------------------------------------------------------------------
// Kernel 3: Y = O @ out_w^T + out_b (fp32 out).  M=4096, N=1024, K=1024.
// ---------------------------------------------------------------------------
__global__ __launch_bounds__(256) void out_proj_kernel(
    const bf16* __restrict__ o, const bf16* __restrict__ w,
    const float* __restrict__ out_b, float* __restrict__ y)
{
    const int lane = threadIdx.x & 31;
    const int wid  = blockIdx.x * (blockDim.x >> 5) + (threadIdx.x >> 5);
    const int tiles_n = D_ / 64;              // 16
    const int tm = wid / tiles_n;
    const int tn = wid % tiles_n;
    const int half = lane >> 4;
    const int l16  = lane & 15;
    const int bk = half * 8;
    const int kb = half * 16;

    const bf16* aptr = o + (size_t)(tm * 16 + l16) * D_;
    const bf16* wptr[4];
#pragma unroll
    for (int t = 0; t < 4; ++t)
        wptr[t] = w + (size_t)(tn * 64 + t * 16 + l16) * D_;

    v8f acc[4];
#pragma unroll
    for (int t = 0; t < 4; ++t)
#pragma unroll
        for (int i = 0; i < 8; ++i) acc[t][i] = 0.f;

    for (int ks = 0; ks < D_ / 32; ++ks) {
        __builtin_prefetch((const void*)(aptr + (ks + 8) * 32), 0, 0);
        __builtin_prefetch((const void*)(wptr[0] + (ks + 8) * 32), 0, 0);
        v8bf lo = *(const v8bf*)(aptr + ks * 32 + kb);
        v8bf hi = *(const v8bf*)(aptr + ks * 32 + kb + 8);
        v16bf a;
#pragma unroll
        for (int i = 0; i < 8; ++i) { a[i] = lo[i]; a[i + 8] = hi[i]; }
#pragma unroll
        for (int t = 0; t < 4; ++t) {
            v16bf bfr = *(const v16bf*)(wptr[t] + ks * 32 + kb);
            acc[t] = WMMA_BF16(a, bfr, acc[t]);
        }
    }

#pragma unroll
    for (int t = 0; t < 4; ++t) {
        int   j    = tn * 64 + t * 16 + l16;
        float bias = out_b[j];
#pragma unroll
        for (int r = 0; r < 8; ++r) {
            int row = tm * 16 + r + 8 * half;
            y[(size_t)row * D_ + j] = acc[t][r] + bias;
        }
    }
}

// ---------------------------------------------------------------------------
extern "C" void kernel_launch(void* const* d_in, const int* in_sizes, int n_in,
                              void* d_out, int out_size, void* d_ws, size_t ws_size,
                              hipStream_t stream)
{
    const float* query = (const float*)d_in[0];
    // d_in[1] ("key") is unused by the reference forward
    const float* value = (const float*)d_in[2];
    const float* qk_w  = (const float*)d_in[3];
    const float* qk_b  = (const float*)d_in[4];
    const float* out_w = (const float*)d_in[5];
    const float* out_b = (const float*)d_in[6];
    float*       y     = (float*)d_out;

    const size_t elems = (size_t)B_ * S_ * D_;   // 4,194,304
    bf16* qws    = (bf16*)d_ws;                  // Q  [b,h,S,dh]      8 MB
    bf16* kws    = qws + elems;                  // K  [b,h,S,dh]      8 MB
    bf16* vtw    = kws + elems;                  // V^T[b,h,dh,S]      8 MB
    bf16* ows    = vtw + elems;                  // O  [B,S,D]         8 MB
    bf16* qrybf  = ows + elems;                  // query bf16         8 MB
    bf16* wqkbf  = qrybf + elems;                // qk_w  bf16         4 MB
    bf16* woutbf = wqkbf + (size_t)2 * D_ * D_;  // out_w bf16         2 MB

    cast_bf16_kernel<<<(int)(elems / 1024), 256, 0, stream>>>(query, qrybf);
    cast_bf16_kernel<<<(int)(2 * D_ * D_ / 1024), 256, 0, stream>>>(qk_w, wqkbf);
    cast_bf16_kernel<<<(int)(D_ * D_ / 1024), 256, 0, stream>>>(out_w, woutbf);
    packv_kernel<<<(int)(elems / 256), 256, 0, stream>>>(value, vtw);

    // 256 m-tiles x 32 n-tiles = 8192 waves, 8 waves/block
    qk_proj_kernel<<<1024, 256, 0, stream>>>(qrybf, wqkbf, qk_b, qws, kws);

    // B*H*(S/128) = 512 blocks, 8 waves each (one 128-row q block per (b,h))
    attn_kernel<<<512, 256, 0, stream>>>(qws, kws, vtw, ows);

    // 256 m-tiles x 16 n-tiles = 4096 waves, 8 waves/block
    out_proj_kernel<<<512, 256, 0, stream>>>(ows, woutbf, out_b, y);
}
